// ContextCTRNN_6373731467517
// MI455X (gfx1250) — matrix operand order repaired and checked by reading, not compiled, not run on hardware
//
#include <hip/hip_runtime.h>
#include <hip/hip_bf16.h>

typedef __bf16 bf16_t;
typedef __attribute__((ext_vector_type(16))) __bf16 v16bf;
typedef __attribute__((ext_vector_type(8)))  __bf16 v8bf;
typedef __attribute__((ext_vector_type(8)))  float  v8f;
typedef __attribute__((ext_vector_type(4)))  float  v4f;

constexpr int Cdim = 64;
constexpr int Vdim = 16000;
constexpr int Tdim = 64;
constexpr int Bdim = 8;
constexpr int Kdim = Cdim * Cdim;   // 4096
constexpr int Mrows = Tdim * Bdim;  // 512

// ---------------------------------------------------------------------------
// Kernel 1: sequential norm-gated context scan. One wave per batch element,
// lane holds channels (lane, lane+32). Writes ctx_all[t*B+b, 0..63] (f32).
// ---------------------------------------------------------------------------
__global__ __launch_bounds__(256) void ctx_scan_kernel(
    const int* __restrict__ tokens, const float* __restrict__ emb_ctx,
    const float* __restrict__ bm_p, const float* __restrict__ bp_p,
    float* __restrict__ ctx_all)
{
    const int lane = threadIdx.x & 31;
    const int b    = threadIdx.x >> 5;   // 8 waves, one per batch row
    const float bm = bm_p[0];
    const float bp = bp_p[0];
    float c0 = 0.f, c1 = 0.f;            // ctx channels lane, lane+32
    for (int t = 0; t < Tdim; ++t) {
        const int tok = tokens[t * Bdim + b];
        const float* ce = emb_ctx + (long)tok * Cdim;
        const float e0 = ce[lane], e1 = ce[lane + 32];
        float se = e0 * e0 + e1 * e1;    // ||ce||^2 partial
        float sc = c0 * c0 + c1 * c1;    // ||ctx||^2 partial
        #pragma unroll
        for (int m = 16; m > 0; m >>= 1) {
            se += __shfl_xor(se, m, 32);
            sc += __shfl_xor(sc, m, 32);
        }
        const float ne = sqrtf(se), nc = sqrtf(sc);
        float beta = powf(bm * (ne / (ne + nc)), bp);
        beta = fminf(fmaxf(beta, 0.f), 1.f);
        c0 = (1.f - beta) * c0 + beta * e0;
        c1 = (1.f - beta) * c1 + beta * e1;
        float* dst = ctx_all + ((long)t * Bdim + b) * Cdim;
        dst[lane] = c0;
        dst[lane + 32] = c1;
    }
}

// ---------------------------------------------------------------------------
// Kernel 2: build A-matrix rows A[r, c*64+d] = ae[r,c] * ctx[r,d] as bf16
// hi/lo split (hi = rne(f), lo = rne(f - hi)) for split-precision WMMA.
// One block per row r = t*B+b.
// ---------------------------------------------------------------------------
__global__ __launch_bounds__(256) void build_a_kernel(
    const int* __restrict__ tokens, const float* __restrict__ emb_act,
    const float* __restrict__ ctx_all,
    bf16_t* __restrict__ A_hi, bf16_t* __restrict__ A_lo)
{
    __shared__ float ae_s[Cdim];
    __shared__ float cx_s[Cdim];
    const int r   = blockIdx.x;          // r = t*B + b
    const int tid = threadIdx.x;
    const int tok = tokens[r];
    if (tid < Cdim)            ae_s[tid]        = emb_act[(long)tok * Cdim + tid];
    else if (tid < 2 * Cdim)   cx_s[tid - Cdim] = ctx_all[(long)r * Cdim + (tid - Cdim)];
    __syncthreads();
    for (int k = tid; k < Kdim; k += 256) {
        const float f = ae_s[k >> 6] * cx_s[k & 63];
        const bf16_t h = (bf16_t)f;
        const bf16_t l = (bf16_t)(f - (float)h);
        const long idx = (long)r * Kdim + k;
        A_hi[idx] = h;
        A_lo[idx] = l;
    }
}

// ---------------------------------------------------------------------------
// Kernel 3: out[512,16000] = A[512,4096] x W[16000,4096]^T via
// v_wmma_f32_16x16x32_bf16, split-precision (hi*hi + hi*lo + lo*hi).
// Grid: x = 4 M-chunks of 128 rows, y = 250 N-tiles of 64 cols (x-fastest so
// the W tile stays hot in L2 across the 4 M-chunks). 8 waves per WG:
// wave = (n_sub in 0..3) x (m_half in 0..1); each wave owns 16 cols x 64 rows
// = 4 accumulator tiles of 16x16.
// ---------------------------------------------------------------------------
union BF16x16 { v16bf v; v8bf h[2]; };

__global__ __launch_bounds__(256) void gemm_wmma_kernel(
    const float* __restrict__ Wmat,          // [V, K] f32, K-major
    const bf16_t* __restrict__ A_hi,
    const bf16_t* __restrict__ A_lo,
    float* __restrict__ out)                 // [M, V] f32
{
    const int lane   = threadIdx.x & 31;
    const int w      = threadIdx.x >> 5;     // 0..7
    const int n_sub  = w & 3;
    const int m_half = w >> 2;
    const int lan16  = lane & 15;
    const int hi16   = lane >> 4;            // 0 or 1
    const int ncol   = blockIdx.y * 64 + n_sub * 16 + lan16;
    const int m_base = blockIdx.x * 128 + m_half * 64;

    v8f acc[4];
    #pragma unroll
    for (int i = 0; i < 4; ++i) acc[i] = (v8f)(0.f);

    for (int kc = 0; kc < Kdim; kc += 32) {
        // ---- B tile: lane holds W[ncol, kc + hi16*16 + 0..15], f32 -> bf16 hi/lo
        const float* wp = Wmat + (long)ncol * Kdim + kc + hi16 * 16;
        const v4f w0 = *(const v4f*)(wp + 0);
        const v4f w1 = *(const v4f*)(wp + 4);
        const v4f w2 = *(const v4f*)(wp + 8);
        const v4f w3 = *(const v4f*)(wp + 12);
        float wf[16];
        *(v4f*)&wf[0]  = w0; *(v4f*)&wf[4]  = w1;
        *(v4f*)&wf[8]  = w2; *(v4f*)&wf[12] = w3;
        BF16x16 bh, bl;
        #pragma unroll
        for (int i = 0; i < 16; ++i) {
            const bf16_t h = (bf16_t)wf[i];
            bh.v[i] = h;
            bl.v[i] = (bf16_t)(wf[i] - (float)h);
        }

        // ---- A tiles + WMMA: elems 0..15 = K[kc+hi16*16 ..]
        const int khalf = hi16 * 16;
        #pragma unroll
        for (int mi = 0; mi < 4; ++mi) {
            const int row = m_base + mi * 16 + lan16;
            const long base = (long)row * Kdim + kc + khalf;
            BF16x16 ah, al;
            ah.h[0] = *(const v8bf*)(A_hi + base);
            ah.h[1] = *(const v8bf*)(A_hi + base + 8);
            al.h[0] = *(const v8bf*)(A_lo + base);
            al.h[1] = *(const v8bf*)(A_lo + base + 8);
            acc[mi] = __builtin_amdgcn_wmma_f32_16x16x32_bf16(
                false, ah.v, false, bh.v, (short)0, acc[mi], false, false);
            acc[mi] = __builtin_amdgcn_wmma_f32_16x16x32_bf16(
                false, ah.v, false, bl.v, (short)0, acc[mi], false, false);
            acc[mi] = __builtin_amdgcn_wmma_f32_16x16x32_bf16(
                false, al.v, false, bh.v, (short)0, acc[mi], false, false);
        }
    }

    // ---- store: C/D layout — lane col = lane&15, VGPR j -> row j + 8*(lane>=16)
    #pragma unroll
    for (int mi = 0; mi < 4; ++mi) {
        #pragma unroll
        for (int j = 0; j < 8; ++j) {
            const int row = m_base + mi * 16 + j + 8 * hi16;
            out[(long)row * Vdim + ncol] = acc[mi][j];
        }
    }
}

// ---------------------------------------------------------------------------
extern "C" void kernel_launch(void* const* d_in, const int* in_sizes, int n_in,
                              void* d_out, int out_size, void* d_ws, size_t ws_size,
                              hipStream_t stream) {
    const int*   tokens  = (const int*)d_in[0];
    const float* emb_ctx = (const float*)d_in[1];
    const float* emb_act = (const float*)d_in[2];
    const float* W_a     = (const float*)d_in[3];
    const float* bm      = (const float*)d_in[4];
    const float* bp      = (const float*)d_in[5];
    float* out = (float*)d_out;

    // Workspace layout: ctx_all (512*64 f32 = 128 KiB), A_hi / A_lo (4 MiB each)
    char* ws = (char*)d_ws;
    float*  ctx_all = (float*)ws;
    bf16_t* A_hi    = (bf16_t*)(ws + (size_t)Mrows * Cdim * sizeof(float));
    bf16_t* A_lo    = A_hi + (size_t)Mrows * Kdim;

    ctx_scan_kernel<<<1, 256, 0, stream>>>(tokens, emb_ctx, bm, bp, ctx_all);
    build_a_kernel<<<Mrows, 256, 0, stream>>>(tokens, emb_act, ctx_all, A_hi, A_lo);
    gemm_wmma_kernel<<<dim3(4, 250), 256, 0, stream>>>(W_a, A_hi, A_lo, out);
}
